// MixtureOfExperts_41308995453103
// MI455X (gfx1250) — compile-verified, matrix operand order checked by reading, not run on hardware
//
#include <hip/hip_runtime.h>
#include <hip/hip_bf16.h>

// ---------------- types ----------------
typedef __bf16 bf16;
typedef __bf16 v16bf __attribute__((ext_vector_type(16)));
typedef __bf16 v8bf  __attribute__((ext_vector_type(8)));
typedef float  v8f   __attribute__((ext_vector_type(8)));

// ---------------- problem constants ----------------
#define T_TOK   16384   // B*S
#define DIM_D   512
#define DIM_H   1024
#define DIM_O   512
#define NEXP    8
#define TILE_M  32
#define HSTR    1032    // bf16 elems per LDS row (pad 8 -> 516 dwords, %64 = 4)
#define OSTR    516     // f32 elems per sOut row

// ---------------- helpers ----------------
__device__ __forceinline__ float gelu_exact(float v) {
    return 0.5f * v * (1.0f + erff(v * 0.70710678118654752440f));
}

// Load one 16x32 bf16 WMMA fragment chunk for this lane.
// p = row/col base pointer at current k offset; kb = (lane>>4)*8.
// lane holds elements [kb..kb+7] and [kb+16..kb+23] of its row (A) / col (B^T).
__device__ __forceinline__ v16bf ldfrag(const bf16* __restrict__ p, int kb) {
    v8bf lo = *reinterpret_cast<const v8bf*>(p + kb);
    v8bf hi = *reinterpret_cast<const v8bf*>(p + kb + 16);
    v16bf r;
#pragma unroll
    for (int i = 0; i < 8; ++i) { r[i] = lo[i]; r[i + 8] = hi[i]; }
    return r;
}

// ---------------- pre-pass: f32 -> bf16 copy ----------------
__global__ void cvt_bf16_kernel(const float* __restrict__ src,
                                bf16* __restrict__ dst, size_t n) {
    for (size_t i = (size_t)blockIdx.x * blockDim.x + threadIdx.x; i < n;
         i += (size_t)gridDim.x * blockDim.x)
        dst[i] = (bf16)src[i];
}

// src: [E][R][C] f32  ->  dst: [E][C][R] bf16 (transposed per expert)
__global__ void transpose_cvt_kernel(const float* __restrict__ src,
                                     bf16* __restrict__ dst, int R, int C) {
    size_t per = (size_t)R * C;
    size_t n = per * NEXP;
    for (size_t i = (size_t)blockIdx.x * blockDim.x + threadIdx.x; i < n;
         i += (size_t)gridDim.x * blockDim.x) {
        size_t e  = i / per;
        size_t rc = i % per;
        size_t c  = rc / R;
        size_t r  = rc % R;
        dst[i] = (bf16)src[(e * R + r) * C + c];
    }
}

// ---------------- router: dense top-2 softmax gates ----------------
__global__ __launch_bounds__(256) void router_kernel(
    const float* __restrict__ x, const float* __restrict__ gw,
    float* __restrict__ gates) {
    __shared__ float sgw[NEXP][DIM_D];
    int tid = threadIdx.x;
    for (int i = tid; i < NEXP * DIM_D; i += 256)
        sgw[i >> 9][i & 511] = gw[i];
    __syncthreads();

    int t = blockIdx.x * 256 + tid;
    const float* xr = x + (size_t)t * DIM_D;
    float lg[NEXP];
#pragma unroll
    for (int e = 0; e < NEXP; ++e) lg[e] = 0.f;
    for (int d = 0; d < DIM_D; ++d) {
        float xv = xr[d];
#pragma unroll
        for (int e = 0; e < NEXP; ++e) lg[e] += xv * sgw[e][d];
    }
    // top-2 (first occurrence wins ties, like lax.top_k)
    int i0 = 0; float v0 = lg[0];
#pragma unroll
    for (int e = 1; e < NEXP; ++e) if (lg[e] > v0) { v0 = lg[e]; i0 = e; }
    int i1 = -1; float v1 = -3.4e38f;
#pragma unroll
    for (int e = 0; e < NEXP; ++e)
        if (e != i0 && lg[e] > v1) { v1 = lg[e]; i1 = e; }
    float g0 = 1.f / (1.f + expf(v1 - v0));
    float g1 = 1.f - g0;
#pragma unroll
    for (int e = 0; e < NEXP; ++e) {
        float g = (e == i0) ? g0 : ((e == i1) ? g1 : 0.f);
        gates[(size_t)t * NEXP + e] = g;
    }
}

// ---------------- aux loss: KL(uniform || usage), deterministic ----------------
__global__ __launch_bounds__(256) void aux_kernel(const float* __restrict__ gates,
                                                  float* __restrict__ aux_out) {
    __shared__ float part[256];
    __shared__ float us[NEXP];
    int tid = threadIdx.x;
    int e = tid & 7, t0 = tid >> 3;
    float s = 0.f;
    for (int t = t0; t < T_TOK; t += 32) s += gates[(size_t)t * NEXP + e];
    part[tid] = s;
    __syncthreads();
    if (tid < NEXP) {
        float u = 0.f;
        for (int k = 0; k < 32; ++k) u += part[tid + 8 * k];  // fixed order
        us[tid] = u / (float)T_TOK;
    }
    __syncthreads();
    if (tid == 0) {
        float a = 0.f;
        const float uni = 1.0f / NEXP, luni = logf(1.0f / NEXP);
        for (int k = 0; k < NEXP; ++k) a += uni * (luni - logf(us[k]));
        aux_out[0] = a;
    }
}

// ---------------- fused MoE FFN + gated combine + LayerNorm ----------------
// Per wave: BOTH 16-row blocks x 4 column tiles -> each B fragment feeds
// two WMMAs (halves L2 weight traffic vs 1x8 blocking).
__global__ __launch_bounds__(256) void moe_ffn_kernel(
    const bf16* __restrict__ xb,     // [T][D] bf16
    const bf16* __restrict__ w1t,    // [E][H][D] bf16 (transposed)
    const bf16* __restrict__ w2t,    // [E][H][H]
    const bf16* __restrict__ w3t,    // [E][O][H]
    const float* __restrict__ b1, const float* __restrict__ b2,
    const float* __restrict__ b3,
    const float* __restrict__ gates,  // [T][E]
    const float* __restrict__ gamma, const float* __restrict__ beta,
    float* __restrict__ out) {       // [T][O]
    extern __shared__ __align__(16) unsigned char smem[];
    bf16*  sH1   = (bf16*)smem;                         // 32*1032*2 = 66048 B
    bf16*  sH2   = (bf16*)(smem + 66048);               // 66048 B
    float* sGate = (float*)(smem + 2 * 66048);          // 128 B
    float* sOut  = (float*)smem;                        // aliases sH1 (32*516*4)

    const int tid   = threadIdx.x;
    const int wave  = tid >> 5;
    const int lane  = tid & 31;
    const int tbase = blockIdx.x * TILE_M;
    const int half  = lane >> 4;
    const int mn    = lane & 15;       // A: row-in-block, B: col-in-tile
    const int kb    = half * 8;

    // persistent gated output accumulators: 2 rowblocks x 4 col tiles
    v8f acc3a[4], acc3b[4];
#pragma unroll
    for (int j = 0; j < 4; ++j)
#pragma unroll
        for (int i = 0; i < 8; ++i) { acc3a[j][i] = 0.f; acc3b[j][i] = 0.f; }

    const bf16* Ax0  = xb  + (size_t)(tbase + mn) * DIM_D;
    const bf16* Ax1  = xb  + (size_t)(tbase + 16 + mn) * DIM_D;
    const bf16* Ah1a = sH1 + (size_t)mn * HSTR;
    const bf16* Ah1b = sH1 + (size_t)(16 + mn) * HSTR;
    const bf16* Ah2a = sH2 + (size_t)mn * HSTR;
    const bf16* Ah2b = sH2 + (size_t)(16 + mn) * HSTR;

    for (int e = 0; e < NEXP; ++e) {
        if (tid < TILE_M) sGate[tid] = gates[(size_t)(tbase + tid) * NEXP + e];
        __syncthreads();  // prev-expert L3 reads of sH2 done; sGate visible

        // ---- Layer 1: xb[32][512] @ w1t[e] -> gelu -> sH1[32][1024]
        {
            const bf16* W = w1t + (size_t)e * DIM_H * DIM_D;
#pragma unroll
            for (int g = 0; g < 2; ++g) {
                const int col0 = wave * 4 + g * 32;
                v8f acc0[4], acc1[4];
#pragma unroll
                for (int j = 0; j < 4; ++j)
#pragma unroll
                    for (int i = 0; i < 8; ++i) { acc0[j][i] = 0.f; acc1[j][i] = 0.f; }
                for (int kk = 0; kk < DIM_D; kk += 32) {
                    v16bf a0 = ldfrag(Ax0 + kk, kb);
                    v16bf a1 = ldfrag(Ax1 + kk, kb);
#pragma unroll
                    for (int j = 0; j < 4; ++j) {
                        const bf16* bp =
                            W + (size_t)((col0 + j) * 16 + mn) * DIM_D + kk;
                        v16bf b = ldfrag(bp, kb);
                        acc0[j] = __builtin_amdgcn_wmma_f32_16x16x32_bf16(
                            false, a0, false, b, (short)0, acc0[j], false, false);
                        acc1[j] = __builtin_amdgcn_wmma_f32_16x16x32_bf16(
                            false, a1, false, b, (short)0, acc1[j], false, false);
                    }
                }
#pragma unroll
                for (int j = 0; j < 4; ++j) {
                    int ncol = (col0 + j) * 16 + mn;
                    float bias = b1[e * DIM_H + ncol];
#pragma unroll
                    for (int i = 0; i < 8; ++i) {
                        int row = i + 8 * half;
                        sH1[row * HSTR + ncol] =
                            (bf16)gelu_exact(acc0[j][i] + bias);
                        sH1[(row + 16) * HSTR + ncol] =
                            (bf16)gelu_exact(acc1[j][i] + bias);
                    }
                }
            }
        }
        __syncthreads();

        // ---- Layer 2: sH1[32][1024] @ w2t[e] -> gelu -> sH2[32][1024]
        {
            const bf16* W = w2t + (size_t)e * DIM_H * DIM_H;
#pragma unroll
            for (int g = 0; g < 2; ++g) {
                const int col0 = wave * 4 + g * 32;
                v8f acc0[4], acc1[4];
#pragma unroll
                for (int j = 0; j < 4; ++j)
#pragma unroll
                    for (int i = 0; i < 8; ++i) { acc0[j][i] = 0.f; acc1[j][i] = 0.f; }
                for (int kk = 0; kk < DIM_H; kk += 32) {
                    v16bf a0 = ldfrag(Ah1a + kk, kb);
                    v16bf a1 = ldfrag(Ah1b + kk, kb);
#pragma unroll
                    for (int j = 0; j < 4; ++j) {
                        const bf16* bp =
                            W + (size_t)((col0 + j) * 16 + mn) * DIM_H + kk;
                        v16bf b = ldfrag(bp, kb);
                        acc0[j] = __builtin_amdgcn_wmma_f32_16x16x32_bf16(
                            false, a0, false, b, (short)0, acc0[j], false, false);
                        acc1[j] = __builtin_amdgcn_wmma_f32_16x16x32_bf16(
                            false, a1, false, b, (short)0, acc1[j], false, false);
                    }
                }
#pragma unroll
                for (int j = 0; j < 4; ++j) {
                    int ncol = (col0 + j) * 16 + mn;
                    float bias = b2[e * DIM_H + ncol];
#pragma unroll
                    for (int i = 0; i < 8; ++i) {
                        int row = i + 8 * half;
                        sH2[row * HSTR + ncol] =
                            (bf16)gelu_exact(acc0[j][i] + bias);
                        sH2[(row + 16) * HSTR + ncol] =
                            (bf16)gelu_exact(acc1[j][i] + bias);
                    }
                }
            }
        }
        __syncthreads();

        // ---- Layer 3: sH2[32][1024] @ w3t[e]; gate-weighted accumulate
        {
            const bf16* W = w3t + (size_t)e * DIM_O * DIM_H;
            const int col0 = wave * 4;
            v8f acc0[4], acc1[4];
#pragma unroll
            for (int j = 0; j < 4; ++j)
#pragma unroll
                for (int i = 0; i < 8; ++i) { acc0[j][i] = 0.f; acc1[j][i] = 0.f; }
            for (int kk = 0; kk < DIM_H; kk += 32) {
                v16bf a0 = ldfrag(Ah2a + kk, kb);
                v16bf a1 = ldfrag(Ah2b + kk, kb);
#pragma unroll
                for (int j = 0; j < 4; ++j) {
                    const bf16* bp =
                        W + (size_t)((col0 + j) * 16 + mn) * DIM_H + kk;
                    v16bf b = ldfrag(bp, kb);
                    acc0[j] = __builtin_amdgcn_wmma_f32_16x16x32_bf16(
                        false, a0, false, b, (short)0, acc0[j], false, false);
                    acc1[j] = __builtin_amdgcn_wmma_f32_16x16x32_bf16(
                        false, a1, false, b, (short)0, acc1[j], false, false);
                }
            }
#pragma unroll
            for (int j = 0; j < 4; ++j) {
                int ncol = (col0 + j) * 16 + mn;
                float bias = b3[e * DIM_O + ncol];
#pragma unroll
                for (int i = 0; i < 8; ++i) {
                    int row = i + 8 * half;
                    float g0 = sGate[row];
                    float g1 = sGate[row + 16];
                    acc3a[j][i] += g0 * (acc0[j][i] + bias);
                    acc3b[j][i] += g1 * (acc1[j][i] + bias);
                }
            }
        }
    }

    // ---- spill gated accumulators to sOut (aliases sH1) ----
    __syncthreads();  // everyone finished reading sH1/sH2
    {
        const int col0 = wave * 4;
#pragma unroll
        for (int j = 0; j < 4; ++j) {
            int ncol = (col0 + j) * 16 + mn;
#pragma unroll
            for (int i = 0; i < 8; ++i) {
                int row = i + 8 * half;
                sOut[row * OSTR + ncol] = acc3a[j][i];
                sOut[(row + 16) * OSTR + ncol] = acc3b[j][i];
            }
        }
    }
    __syncthreads();

    // ---- LayerNorm: each wave owns 4 rows; shuffle reduction over 32 lanes
#pragma unroll
    for (int rr = 0; rr < 4; ++rr) {
        int row = wave * 4 + rr;
        int t = tbase + row;
        float vals[16];
        float s = 0.f, s2 = 0.f;
#pragma unroll
        for (int i = 0; i < 16; ++i) {
            float v = sOut[row * OSTR + lane + 32 * i];
            vals[i] = v;
            s += v;
            s2 += v * v;
        }
#pragma unroll
        for (int off = 16; off > 0; off >>= 1) {
            s  += __shfl_xor(s, off, 32);
            s2 += __shfl_xor(s2, off, 32);
        }
        float mean = s * (1.f / DIM_O);
        float var  = s2 * (1.f / DIM_O) - mean * mean;
        float inv  = rsqrtf(var + 1e-5f);
#pragma unroll
        for (int i = 0; i < 16; ++i) {
            int c = lane + 32 * i;
            out[(size_t)t * DIM_O + c] =
                (vals[i] - mean) * inv * gamma[c] + beta[c];
        }
    }
}

// ---------------- launcher ----------------
extern "C" void kernel_launch(void* const* d_in, const int* in_sizes, int n_in,
                              void* d_out, int out_size, void* d_ws,
                              size_t ws_size, hipStream_t stream) {
    const float* x     = (const float*)d_in[0];
    const float* gw    = (const float*)d_in[1];
    const float* w1    = (const float*)d_in[2];
    const float* b1    = (const float*)d_in[3];
    const float* w2    = (const float*)d_in[4];
    const float* b2    = (const float*)d_in[5];
    const float* w3    = (const float*)d_in[6];
    const float* b3    = (const float*)d_in[7];
    const float* gamma = (const float*)d_in[8];
    const float* beta  = (const float*)d_in[9];
    float* out = (float*)d_out;

    // workspace layout (bytes)
    char* ws = (char*)d_ws;
    float* gates = (float*)ws;                       // T*8*4      = 524288
    bf16* xb  = (bf16*)(ws + 524288);                // T*512*2    = 16777216
    bf16* w1t = (bf16*)(ws + 524288 + 16777216);     // 8*1024*512*2 = 8388608
    bf16* w2t = (bf16*)(ws + 524288 + 16777216 + 8388608);   // 16777216
    bf16* w3t = (bf16*)(ws + 524288 + 16777216 + 8388608 + 16777216); // 8388608

    cvt_bf16_kernel<<<4096, 256, 0, stream>>>(x, xb, (size_t)T_TOK * DIM_D);
    transpose_cvt_kernel<<<4096, 256, 0, stream>>>(w1, w1t, DIM_D, DIM_H);
    transpose_cvt_kernel<<<4096, 256, 0, stream>>>(w2, w2t, DIM_H, DIM_H);
    transpose_cvt_kernel<<<4096, 256, 0, stream>>>(w3, w3t, DIM_H, DIM_O);

    router_kernel<<<T_TOK / 256, 256, 0, stream>>>(x, gw, gates);
    aux_kernel<<<1, 256, 0, stream>>>(gates, out + (size_t)T_TOK * DIM_O);

    const size_t lds_bytes = 2 * 66048 + 128;  // sH1 + sH2 + sGate
    moe_ffn_kernel<<<T_TOK / TILE_M, 256, lds_bytes, stream>>>(
        xb, w1t, w2t, w3t, b1, b2, b3, gates, gamma, beta, out);
}
